// MultiHeadSelfAttention_72997264163054
// MI455X (gfx1250) — compile-verified, matrix operand order checked by reading, not run on hardware
//
#include <hip/hip_runtime.h>
#include <hip/hip_bf16.h>

#define HIDDEN 512
#define HEADS  8
#define HEADD  64
#define BATCH  2
#define SEQ    4096
#define TOKENS (BATCH * SEQ)

typedef __attribute__((ext_vector_type(16))) __bf16 v16bf;
typedef __attribute__((ext_vector_type(8)))  __bf16 v8bf;
typedef __attribute__((ext_vector_type(8)))  float  v8f;

// ---------- helpers ----------
__device__ __forceinline__ unsigned short bf_bits(float f) {
  union { float f; unsigned u; } c; c.f = f;
  unsigned u = c.u + 0x7FFFu + ((c.u >> 16) & 1u);   // round-to-nearest-even
  return (unsigned short)(u >> 16);
}
__device__ __forceinline__ __bf16 f2bf(float f) {
  union { unsigned short s; __bf16 b; } o; o.s = bf_bits(f);
  return o.b;
}
// cheap truncating conversion for P in [0,1] (hot path)
__device__ __forceinline__ __bf16 f2bf_trunc(float f) {
  union { float f; unsigned u; } c; c.f = f;
  union { unsigned short s; __bf16 b; } o; o.s = (unsigned short)(c.u >> 16);
  return o.b;
}

__device__ __forceinline__ v8f wmma_bf16(v16bf a, v16bf b, v8f c) {
  // D = A(16x32) * B(32x16) + C, fp32 accum
  return __builtin_amdgcn_wmma_f32_16x16x32_bf16(
      false, a, false, b, (short)0, c, false, false);
}

// Load a 16x32 bf16 operand tile (rows contiguous with leading dim ld) into
// the documented 16-bit A-matrix VGPR layout (B assumed symmetric in N):
// lane = row%16, half=lane>>4; VGPR i holds K-pair at (i&4)<<2 | half<<3 | (i&3)<<1
// The 4 pairs per K-halfwindow are 16B contiguous -> b128 loads.
__device__ __forceinline__ v16bf load_tile_bf(const __bf16* p, int ld) {
  const int lane = threadIdx.x & 31;
  const __bf16* row = p + (size_t)(lane & 15) * ld;
  const int half = (lane >> 4) & 1;
  union { v16bf v; unsigned u[8]; } o;
#pragma unroll
  for (int i = 0; i < 8; ++i) {
    const int kb = ((i & 4) << 2) + (half << 3) + ((i & 3) << 1);
    o.u[i] = *(const unsigned*)(row + kb);          // packed bf16 pair
  }
  return o.v;
}

// ---------- kernel 0: one-shot fp32 -> bf16 conversion (bandwidth-bound) ----------
__global__ __launch_bounds__(256) void f32_to_bf16_kernel(
    const float* __restrict__ src, __bf16* __restrict__ dst, int n) {
  const int i = (blockIdx.x * 256 + threadIdx.x) * 8;
  if (i >= n) return;
  const float4 a = *(const float4*)(src + i);
  const float4 b = *(const float4*)(src + i + 4);
  union { v8bf v; __bf16 e[8]; } o;
  o.e[0] = f2bf(a.x); o.e[1] = f2bf(a.y); o.e[2] = f2bf(a.z); o.e[3] = f2bf(a.w);
  o.e[4] = f2bf(b.x); o.e[5] = f2bf(b.y); o.e[6] = f2bf(b.z); o.e[7] = f2bf(b.w);
  *(v8bf*)(dst + i) = o.v;
}

// ---------- kernel 1: fused Q/K/V projections (X @ W^T + b), all-bf16 tiles ----------
// Q,K stored [B,H,S,D]; V stored transposed [B,H,D,S]. Q is pre-scaled by
// 1/sqrt(HEADD) so the attention kernel needs no per-logit scaling.
__global__ __launch_bounds__(256) void qkv_proj_kernel(
    const __bf16* __restrict__ xq, const __bf16* __restrict__ xk,
    const __bf16* __restrict__ xv,
    const __bf16* __restrict__ wq, const __bf16* __restrict__ wk,
    const __bf16* __restrict__ wv,
    const float* __restrict__ qbias, const float* __restrict__ kbias,
    const float* __restrict__ vbias,
    __bf16* __restrict__ Qb, __bf16* __restrict__ Kb, __bf16* __restrict__ Vtb) {
  const int wid  = blockIdx.x * 8 + (threadIdx.x >> 5);
  const int proj = wid >> 12;            // 512*8 = 4096 wave-tiles per projection
  const int rem  = wid & 4095;
  const int m0   = (rem >> 3) * 16;      // token tile
  const int n0   = (rem & 7) * 64;       // 64-wide output strip

  const __bf16* X  = proj == 0 ? xq : (proj == 1 ? xk : xv);
  const __bf16* W  = proj == 0 ? wq : (proj == 1 ? wk : wv);
  const float*  Bv = proj == 0 ? qbias : (proj == 1 ? kbias : vbias);
  const float scale = (proj == 0) ? 0.125f : 1.0f;   // 1/sqrt(64) folded into Q

  v8f acc[4];
#pragma unroll
  for (int j = 0; j < 4; ++j)
#pragma unroll
    for (int e = 0; e < 8; ++e) acc[j][e] = 0.f;

#pragma unroll 1
  for (int kk = 0; kk < HIDDEN; kk += 32) {
    const v16bf a = load_tile_bf(X + (size_t)m0 * HIDDEN + kk, HIDDEN);
#pragma unroll
    for (int j = 0; j < 4; ++j) {
      const v16bf b = load_tile_bf(W + (size_t)(n0 + 16 * j) * HIDDEN + kk, HIDDEN);
      acc[j] = wmma_bf16(a, b, acc[j]);
    }
  }

  const int lane = threadIdx.x & 31;
  const int n  = lane & 15;
  const int mh = (lane >> 4) << 3;       // +8 rows for upper lane half
#pragma unroll
  for (int j = 0; j < 4; ++j) {
    const int feat = n0 + 16 * j + n;
    const float bias = Bv[feat];
    const int h = feat >> 6, d = feat & 63;
#pragma unroll
    for (int v = 0; v < 8; ++v) {
      const int tok = m0 + v + mh;
      const int b_  = tok >> 12;         // /SEQ
      const int s   = tok & (SEQ - 1);
      const float val = (acc[j][v] + bias) * scale;
      if (proj == 2)
        Vtb[(((size_t)b_ * HEADS + h) * HEADD + d) * SEQ + s] = f2bf(val);
      else {
        __bf16* dst = (proj == 0) ? Qb : Kb;
        dst[(((size_t)b_ * HEADS + h) * SEQ + s) * HEADD + d] = f2bf(val);
      }
    }
  }
}

// ---------- kernel 2: flash attention, one 16-query tile per wave, 64-token steps ----------
__global__ __launch_bounds__(256) void flash_attn_kernel(
    const __bf16* __restrict__ Qb, const __bf16* __restrict__ Kb,
    const __bf16* __restrict__ Vtb, const unsigned char* __restrict__ mask,
    __bf16* __restrict__ ctx) {
  __shared__ __bf16 lds[8][16][64];      // per-wave P staging (C-layout -> A-layout)
  const int w    = threadIdx.x >> 5;
  const int lane = threadIdx.x & 31;
  const int n    = lane & 15;
  const int half = lane >> 4;
  const int bh   = blockIdx.x >> 5;      // 32 q-blocks of 128 per (b,h)
  const int qblk = blockIdx.x & 31;
  const int b    = bh >> 3;
  const int h    = bh & 7;
  const int q0   = qblk * 128 + w * 16;

  const __bf16* Qh = Qb + ((size_t)bh * SEQ + q0) * HEADD;
  const __bf16* Kh = Kb + (size_t)bh * SEQ * HEADD;
  const __bf16* Vh = Vtb + (size_t)bh * HEADD * SEQ;
  const unsigned char* Mh = mask + (size_t)b * SEQ * SEQ;

  const v16bf qa0 = load_tile_bf(Qh, HEADD);        // d = 0..31 (Q pre-scaled)
  const v16bf qa1 = load_tile_bf(Qh + 32, HEADD);   // d = 32..63

  v8f o[4];
  float mrun[8], lsum[8];                // lsum = per-lane partial row sum
#pragma unroll
  for (int v = 0; v < 8; ++v) { mrun[v] = -INFINITY; lsum[v] = 0.f; }
#pragma unroll
  for (int j = 0; j < 4; ++j)
#pragma unroll
    for (int e = 0; e < 8; ++e) o[j][e] = 0.f;

#pragma unroll 1
  for (int k0 = 0; k0 < SEQ; k0 += 64) {
    const __bf16* Kt = Kh + (size_t)k0 * HEADD;
    if (k0 + 64 < SEQ) {                 // global_prefetch_b8 of next K/V stream
      __builtin_prefetch(Kt + 64 * HEADD + (size_t)lane * HEADD, 0, 1);
      __builtin_prefetch(Vh + (size_t)lane * SEQ + k0 + 64, 0, 1);
    }

    // logits: 4 token-subtiles x (2 x K=32) = 8 WMMA
    v8f s[4];
#pragma unroll
    for (int j = 0; j < 4; ++j) {
#pragma unroll
      for (int e = 0; e < 8; ++e) s[j][e] = 0.f;
      const __bf16* kt = Kt + (size_t)(16 * j) * HEADD;
      s[j] = wmma_bf16(qa0, load_tile_bf(kt, HEADD), s[j]);
      s[j] = wmma_bf16(qa1, load_tile_bf(kt + 32, HEADD), s[j]);
    }

    // masked online softmax; row max reduced now, row sum deferred (per-lane partials)
#pragma unroll
    for (int v = 0; v < 8; ++v) {
      const int m = v + half * 8;
      const unsigned char* mp = Mh + (size_t)(q0 + m) * SEQ + k0;
      const float x0 = mp[n]      ? -1e9f : s[0][v];
      const float x1 = mp[n + 16] ? -1e9f : s[1][v];
      const float x2 = mp[n + 32] ? -1e9f : s[2][v];
      const float x3 = mp[n + 48] ? -1e9f : s[3][v];
      float rm = fmaxf(fmaxf(x0, x1), fmaxf(x2, x3));
      rm = fmaxf(rm, __shfl_xor(rm, 1, 16));
      rm = fmaxf(rm, __shfl_xor(rm, 2, 16));
      rm = fmaxf(rm, __shfl_xor(rm, 4, 16));
      rm = fmaxf(rm, __shfl_xor(rm, 8, 16));
      const float mnew = fmaxf(mrun[v], rm);
      const float corr = __expf(mrun[v] - mnew);
      mrun[v] = mnew;
      const float e0 = __expf(x0 - mnew);
      const float e1 = __expf(x1 - mnew);
      const float e2 = __expf(x2 - mnew);
      const float e3 = __expf(x3 - mnew);
      lsum[v] = lsum[v] * corr + ((e0 + e1) + (e2 + e3));
      o[0][v] *= corr; o[1][v] *= corr; o[2][v] *= corr; o[3][v] *= corr;
      lds[w][m][n]      = f2bf_trunc(e0);
      lds[w][m][n + 16] = f2bf_trunc(e1);
      lds[w][m][n + 32] = f2bf_trunc(e2);
      lds[w][m][n + 48] = f2bf_trunc(e3);
    }
    asm volatile("s_wait_dscnt 0" ::: "memory");   // DS in-order per wave; stores done
    const v16bf pa0 = load_tile_bf(&lds[w][0][0], 64);        // k = 0..31
    const v16bf pa1 = load_tile_bf(&lds[w][0][0] + 32, 64);   // k = 32..63

    // O += P * V : 4 d-subtiles x 2 K-steps = 8 WMMA
#pragma unroll
    for (int j = 0; j < 4; ++j) {
      const __bf16* vt = Vh + (size_t)(16 * j) * SEQ + k0;
      o[j] = wmma_bf16(pa0, load_tile_bf(vt, SEQ), o[j]);
      o[j] = wmma_bf16(pa1, load_tile_bf(vt + 32, SEQ), o[j]);
    }
  }

  // finalize: reduce the deferred per-lane partial sums once, then normalize
#pragma unroll
  for (int v = 0; v < 8; ++v) {
    float l = lsum[v];
    l += __shfl_xor(l, 1, 16);
    l += __shfl_xor(l, 2, 16);
    l += __shfl_xor(l, 4, 16);
    l += __shfl_xor(l, 8, 16);
    const float inv = 1.f / l;
#pragma unroll
    for (int j = 0; j < 4; ++j) o[j][v] *= inv;
  }
#pragma unroll
  for (int j = 0; j < 4; ++j) {
    const int feat = h * HEADD + 16 * j + n;
#pragma unroll
    for (int v = 0; v < 8; ++v) {
      const int tok = b * SEQ + q0 + v + half * 8;
      ctx[(size_t)tok * HIDDEN + feat] = f2bf(o[j][v]);
    }
  }
}

// ---------- kernel 3: output projection (ctx @ o_w^T + o_b), fp32 out ----------
__global__ __launch_bounds__(256) void out_proj_kernel(
    const __bf16* __restrict__ ctx, const __bf16* __restrict__ ow,
    const float* __restrict__ ob, float* __restrict__ out) {
  const int wid = blockIdx.x * 8 + (threadIdx.x >> 5);
  const int m0  = (wid >> 3) * 16;
  const int n0  = (wid & 7) * 64;

  v8f acc[4];
#pragma unroll
  for (int j = 0; j < 4; ++j)
#pragma unroll
    for (int e = 0; e < 8; ++e) acc[j][e] = 0.f;

#pragma unroll 1
  for (int kk = 0; kk < HIDDEN; kk += 32) {
    const v16bf a = load_tile_bf(ctx + (size_t)m0 * HIDDEN + kk, HIDDEN);
#pragma unroll
    for (int j = 0; j < 4; ++j) {
      const v16bf b = load_tile_bf(ow + (size_t)(n0 + 16 * j) * HIDDEN + kk, HIDDEN);
      acc[j] = wmma_bf16(a, b, acc[j]);
    }
  }

  const int lane = threadIdx.x & 31;
  const int n  = lane & 15;
  const int mh = (lane >> 4) << 3;
#pragma unroll
  for (int j = 0; j < 4; ++j) {
    const int feat = n0 + 16 * j + n;
    const float bias = ob[feat];
#pragma unroll
    for (int v = 0; v < 8; ++v)
      out[(size_t)(m0 + v + mh) * HIDDEN + feat] = acc[j][v] + bias;
  }
}

// ---------- launcher ----------
extern "C" void kernel_launch(void* const* d_in, const int* in_sizes, int n_in,
                              void* d_out, int out_size, void* d_ws, size_t ws_size,
                              hipStream_t stream) {
  const float* qt = (const float*)d_in[0];
  const float* kt = (const float*)d_in[1];
  const float* vt = (const float*)d_in[2];
  const unsigned char* mask = (const unsigned char*)d_in[3];  // jnp.bool_ = 1 byte
  const float* qw = (const float*)d_in[4];
  const float* qb = (const float*)d_in[5];
  const float* kw = (const float*)d_in[6];
  const float* kb = (const float*)d_in[7];
  const float* vw = (const float*)d_in[8];
  const float* vb = (const float*)d_in[9];
  const float* ow = (const float*)d_in[10];
  const float* ob = (const float*)d_in[11];

  // workspace layout (bf16 staging):
  //   Xq/Xk/Xv : 3 x 8 MiB, Wq/Wk/Wv/Wo : 4 x 512 KiB,
  //   Q / K / V^T / ctx : 4 x 8 MiB                       total ~58 MiB
  char* ws = (char*)d_ws;
  const size_t XSZ = (size_t)TOKENS * HIDDEN * sizeof(__bf16);   // 8 MiB
  const size_t WSZ = (size_t)HIDDEN * HIDDEN * sizeof(__bf16);   // 512 KiB
  __bf16* Xq  = (__bf16*)(ws);
  __bf16* Xk  = (__bf16*)(ws + XSZ);
  __bf16* Xv  = (__bf16*)(ws + 2 * XSZ);
  __bf16* Wq  = (__bf16*)(ws + 3 * XSZ);
  __bf16* Wk  = (__bf16*)(ws + 3 * XSZ + WSZ);
  __bf16* Wv  = (__bf16*)(ws + 3 * XSZ + 2 * WSZ);
  __bf16* Wo  = (__bf16*)(ws + 3 * XSZ + 3 * WSZ);
  __bf16* Qb  = (__bf16*)(ws + 3 * XSZ + 4 * WSZ);
  __bf16* Kb  = (__bf16*)(ws + 4 * XSZ + 4 * WSZ);
  __bf16* Vtb = (__bf16*)(ws + 5 * XSZ + 4 * WSZ);
  __bf16* ctx = (__bf16*)(ws + 6 * XSZ + 4 * WSZ);

  const int NX = TOKENS * HIDDEN;   // 4,194,304
  const int NW = HIDDEN * HIDDEN;   //   262,144
  f32_to_bf16_kernel<<<NX / 2048, 256, 0, stream>>>(qt, Xq, NX);
  f32_to_bf16_kernel<<<NX / 2048, 256, 0, stream>>>(kt, Xk, NX);
  f32_to_bf16_kernel<<<NX / 2048, 256, 0, stream>>>(vt, Xv, NX);
  f32_to_bf16_kernel<<<NW / 2048, 256, 0, stream>>>(qw, Wq, NW);
  f32_to_bf16_kernel<<<NW / 2048, 256, 0, stream>>>(kw, Wk, NW);
  f32_to_bf16_kernel<<<NW / 2048, 256, 0, stream>>>(vw, Wv, NW);
  f32_to_bf16_kernel<<<NW / 2048, 256, 0, stream>>>(ow, Wo, NW);

  // 3 projections * 512 m-tiles * 8 n-strips = 12288 wave-tiles / 8 waves per block
  qkv_proj_kernel<<<1536, 256, 0, stream>>>(Xq, Xk, Xv, Wq, Wk, Wv,
                                            qb, kb, vb, Qb, Kb, Vtb);
  // B*H*(S/128) = 512 blocks
  flash_attn_kernel<<<512, 256, 0, stream>>>(Qb, Kb, Vtb, mask, ctx);
  // 512 m-tiles * 8 n-strips / 8 waves
  out_proj_kernel<<<512, 256, 0, stream>>>(ctx, Wo, ob, (float*)d_out);
}
